// TopkLinear_9792525435094
// MI455X (gfx1250) — matrix operand order repaired and checked by reading, not compiled, not run on hardware
//
#include <hip/hip_runtime.h>

// ---------------- workspace layout (as unsigned words) ----------------
#define WS_PREFIX 0   // current radix-select key prefix (high bits resolved)
#define WS_RANK   1   // remaining rank within current prefix
#define WS_CNTEQ  2   // count of elements equal to selected key (after pass 3)
#define WS_MIN    3   // min key strictly above selected key
#define WS_THR    4   // final threshold (float bits)
#define WS_BINS   8   // 256 histogram bins

#define NUMEL   67108864ull          // 8192*8192
#define K_RANK  3355442u             // floor(p*(N-1)), p = 3355443/N
#define IN_F    8192
#define OUT_F   8192
#define BATCH   512

typedef __attribute__((ext_vector_type(8)))  float  v8f;
typedef __attribute__((ext_vector_type(8)))  __bf16 v8bf;
typedef __attribute__((ext_vector_type(16))) __bf16 v16bf;

__device__ __forceinline__ unsigned f2key(float x) {
  unsigned b = __float_as_uint(x);
  return b ^ ((b & 0x80000000u) ? 0xFFFFFFFFu : 0x80000000u);
}
__device__ __forceinline__ float key2f(unsigned k) {
  unsigned b = (k & 0x80000000u) ? (k ^ 0x80000000u) : ~k;
  return __uint_as_float(b);
}

// bf16 helpers (round-to-nearest-even; inputs are finite)
__device__ __forceinline__ unsigned f2bf(float x) {
  unsigned u = __float_as_uint(x);
  u += 0x7FFFu + ((u >> 16) & 1u);
  return u >> 16;
}
__device__ __forceinline__ float bf2f(unsigned b) {
  return __uint_as_float(b << 16);
}

// ---------------- quantile: init ----------------
__global__ __launch_bounds__(256) void qk_init(unsigned* __restrict__ ws) {
  ws[WS_BINS + threadIdx.x] = 0u;
  if (threadIdx.x == 0) {
    ws[WS_PREFIX] = 0u;
    ws[WS_RANK]   = K_RANK;
    ws[WS_CNTEQ]  = 0u;
    ws[WS_MIN]    = 0xFFFFFFFFu;
    ws[WS_THR]    = 0u;
  }
}

// ---------------- quantile: histogram pass (templated on radix pass) ----------------
template <int PASS>
__global__ __launch_bounds__(256) void qk_hist(const float4* __restrict__ w4,
                                               unsigned* __restrict__ ws) {
  __shared__ unsigned bins[256];
  bins[threadIdx.x] = 0u;
  __syncthreads();

  constexpr int SHIFT = 24 - PASS * 8;
  constexpr int HS    = (PASS == 0) ? 0 : (SHIFT + 8);
  unsigned pref = 0u;
  if (PASS > 0) pref = ws[WS_PREFIX] >> HS;

  const size_t n4     = NUMEL / 4;
  const size_t stride = (size_t)gridDim.x * 256u;
  for (size_t i = (size_t)blockIdx.x * 256u + threadIdx.x; i < n4; i += stride) {
    float4 v = w4[i];
    float e[4] = {v.x, v.y, v.z, v.w};
#pragma unroll
    for (int j = 0; j < 4; ++j) {
      unsigned key = f2key(e[j]);
      bool match = (PASS == 0) || ((key >> HS) == pref);
      if (match) atomicAdd(&bins[(key >> SHIFT) & 255u], 1u);
    }
  }
  __syncthreads();
  unsigned c = bins[threadIdx.x];
  if (c) atomicAdd(&ws[WS_BINS + threadIdx.x], c);
}

// ---------------- quantile: pick bin, advance prefix, clear bins ----------------
__global__ __launch_bounds__(256) void qk_select(unsigned* __restrict__ ws, int pass) {
  if (threadIdx.x == 0) {
    unsigned r = ws[WS_RANK];
    unsigned cum = 0u, cnt = 0u;
    int b = 255;
    for (int i = 0; i < 256; ++i) {
      cnt = ws[WS_BINS + i];
      if (cum + cnt > r) { b = i; break; }
      cum += cnt;
    }
    int shift = 24 - pass * 8;
    ws[WS_PREFIX] = ws[WS_PREFIX] | ((unsigned)b << shift);
    ws[WS_RANK]   = r - cum;
    if (pass == 3) ws[WS_CNTEQ] = cnt;
  }
  __syncthreads();
  ws[WS_BINS + threadIdx.x] = 0u;   // ready for next pass
}

// ---------------- quantile: min key strictly above selected key ----------------
__global__ __launch_bounds__(256) void qk_minabove(const float4* __restrict__ w4,
                                                   unsigned* __restrict__ ws) {
  const unsigned keyk = ws[WS_PREFIX];
  unsigned lmin = 0xFFFFFFFFu;
  const size_t n4     = NUMEL / 4;
  const size_t stride = (size_t)gridDim.x * 256u;
  for (size_t i = (size_t)blockIdx.x * 256u + threadIdx.x; i < n4; i += stride) {
    float4 v = w4[i];
    float e[4] = {v.x, v.y, v.z, v.w};
#pragma unroll
    for (int j = 0; j < 4; ++j) {
      unsigned key = f2key(e[j]);
      if (key > keyk && key < lmin) lmin = key;
    }
  }
  if (lmin != 0xFFFFFFFFu) atomicMin(&ws[WS_MIN], lmin);
}

// ---------------- quantile: finalize thr with linear interpolation ----------------
__global__ void qk_finalize(unsigned* __restrict__ ws) {
  unsigned keyk  = ws[WS_PREFIX];
  unsigned r     = ws[WS_RANK];
  unsigned cnteq = ws[WS_CNTEQ];
  float vk  = key2f(keyk);
  float vk1 = (r + 1u < cnteq) ? vk : key2f(ws[WS_MIN]);
  double q    = 3355443.0 / 67108864.0;
  double idx  = q * (double)(NUMEL - 1ull);
  double frac = idx - (double)K_RANK;
  float thr = (float)((double)vk + frac * ((double)vk1 - (double)vk));
  ws[WS_THR] = __float_as_uint(thr);
}

// ---------------- fused masked GEMM via bf16x3 split ----------------
// C = A @ mask(W)^T + bias.  A[512,8192] f32, W[8192,8192] f32.
// Each f32 operand x is split as x = hi + lo (bf16 each); per 16x16x32 step:
//   acc += aH*bH + aH*bL + aL*bH   (error ~2^-18 per product, fp32-class result)
// Workgroup tile 128(M)x128(N), K-strip 32. 8 waves, wave tile 64x32
// = 4x2 subtiles, 3 x V_WMMA_F32_16X16X32_BF16 per subtile per strip.
#define TM 128
#define TN 128
#define TK 32
#define STR 40   // padded LDS row stride in bf16 elems (80B=20 banks; conflict-free)

__device__ __forceinline__ v16bf ld_op(const unsigned short* p, int second_off) {
  v8bf a = *(const v8bf*)p;
  v8bf b = *(const v8bf*)(p + second_off);
  return __builtin_shufflevector(a, b, 0, 1, 2, 3, 4, 5, 6, 7,
                                 8, 9, 10, 11, 12, 13, 14, 15);
}

__global__ __launch_bounds__(256) void gemm_topk(
    const float* __restrict__ A, const float* __restrict__ W,
    const float* __restrict__ bias, const unsigned* __restrict__ ws,
    float* __restrict__ C) {
  __shared__ unsigned short sAh[TM * STR];
  __shared__ unsigned short sAl[TM * STR];
  __shared__ unsigned short sWh[TN * STR];
  __shared__ unsigned short sWl[TN * STR];

  const float thr = __uint_as_float(ws[WS_THR]);
  const int tid  = threadIdx.x;
  const int wave = tid >> 5;
  const int lane = tid & 31;
  const int m0 = blockIdx.y * TM;
  const int n0 = blockIdx.x * TN;
  const int mw = (wave & 1) * 64;   // wave M origin within tile
  const int nw = (wave >> 1) * 32;  // wave N origin within tile
  const int lm = lane & 15;
  const int h  = lane >> 4;

  v8f acc[4][2];
#pragma unroll
  for (int i = 0; i < 4; ++i)
#pragma unroll
    for (int j = 0; j < 2; ++j)
      acc[i][j] = (v8f){0.f, 0.f, 0.f, 0.f, 0.f, 0.f, 0.f, 0.f};

  for (int k0 = 0; k0 < IN_F; k0 += TK) {
    // ---- stage: load f32 tiles, mask W, split into bf16 hi/lo, store to LDS ----
#pragma unroll
    for (int j = 0; j < 4; ++j) {
      int f   = tid + j * 256;     // 0..1023 float4 slots
      int row = f >> 3;            // 8 float4 per 32-float row
      int c4  = (f & 7) * 4;
      float4 av = *(const float4*)(A + (size_t)(m0 + row) * IN_F + k0 + c4);
      float4 wv = *(const float4*)(W + (size_t)(n0 + row) * IN_F + k0 + c4);
      wv.x = (wv.x <= thr) ? wv.x : 0.0f;
      wv.y = (wv.y <= thr) ? wv.y : 0.0f;
      wv.z = (wv.z <= thr) ? wv.z : 0.0f;
      wv.w = (wv.w <= thr) ? wv.w : 0.0f;
      float ae[4] = {av.x, av.y, av.z, av.w};
      float we[4] = {wv.x, wv.y, wv.z, wv.w};
      uint2 ah, al, wh, wl;
      unsigned aht[4], alt[4], wht[4], wlt[4];
#pragma unroll
      for (int e = 0; e < 4; ++e) {
        unsigned hA = f2bf(ae[e]); aht[e] = hA; alt[e] = f2bf(ae[e] - bf2f(hA));
        unsigned hW = f2bf(we[e]); wht[e] = hW; wlt[e] = f2bf(we[e] - bf2f(hW));
      }
      ah.x = aht[0] | (aht[1] << 16); ah.y = aht[2] | (aht[3] << 16);
      al.x = alt[0] | (alt[1] << 16); al.y = alt[2] | (alt[3] << 16);
      wh.x = wht[0] | (wht[1] << 16); wh.y = wht[2] | (wht[3] << 16);
      wl.x = wlt[0] | (wlt[1] << 16); wl.y = wlt[2] | (wlt[3] << 16);
      const int o = row * STR + c4;
      *(uint2*)(sAh + o) = ah;
      *(uint2*)(sAl + o) = al;
      *(uint2*)(sWh + o) = wh;
      *(uint2*)(sWl + o) = wl;
    }
    __syncthreads();

    // ---- compute: one 16x16x32 K-step, 3 WMMAs per subtile (bf16x3) ----
    // A operand (16x32 bf16): lane holds K = h*8+[0..7] and 16+h*8+[0..7]
    // B operand (32x16 bf16): lane holds K = h*8+[0..7] and 16+h*8+[0..7] of column n=lm
    v16bf aH[4], aL[4], bH[2], bL[2];
#pragma unroll
    for (int i = 0; i < 4; ++i) {
      const int o = (mw + i * 16 + lm) * STR + h * 8;
      aH[i] = ld_op(sAh + o, 16);
      aL[i] = ld_op(sAl + o, 16);
    }
#pragma unroll
    for (int j = 0; j < 2; ++j) {
      const int o = (nw + j * 16 + lm) * STR + h * 8;
      bH[j] = ld_op(sWh + o, 16);
      bL[j] = ld_op(sWl + o, 16);
    }
#pragma unroll
    for (int i = 0; i < 4; ++i)
#pragma unroll
      for (int j = 0; j < 2; ++j) {
        acc[i][j] = __builtin_amdgcn_wmma_f32_16x16x32_bf16(
            false, aH[i], false, bH[j], (short)0, acc[i][j], false, false);
        acc[i][j] = __builtin_amdgcn_wmma_f32_16x16x32_bf16(
            false, aH[i], false, bL[j], (short)0, acc[i][j], false, false);
        acc[i][j] = __builtin_amdgcn_wmma_f32_16x16x32_bf16(
            false, aL[i], false, bH[j], (short)0, acc[i][j], false, false);
      }
    __syncthreads();
  }

  // epilogue: C layout — VGPR v holds M=v (lanes 0-15) / M=v+8 (lanes 16-31), N=lane&15
#pragma unroll
  for (int j = 0; j < 2; ++j) {
    const int n  = n0 + nw + j * 16 + lm;
    const float bv = bias[n];
#pragma unroll
    for (int i = 0; i < 4; ++i) {
      const int mbase = m0 + mw + i * 16 + h * 8;
#pragma unroll
      for (int v = 0; v < 8; ++v)
        C[(size_t)(mbase + v) * OUT_F + n] = acc[i][j][v] + bv;
    }
  }
}

// ---------------- launcher ----------------
extern "C" void kernel_launch(void* const* d_in, const int* in_sizes, int n_in,
                              void* d_out, int out_size, void* d_ws, size_t ws_size,
                              hipStream_t stream) {
  (void)in_sizes; (void)n_in; (void)out_size; (void)ws_size;
  const float* x = (const float*)d_in[0];       // [512, 8192]
  const float* w = (const float*)d_in[1];       // [8192, 8192]
  const float* b = (const float*)d_in[2];       // [8192]
  float* out   = (float*)d_out;                 // [512, 8192]
  unsigned* ws = (unsigned*)d_ws;

  const float4* w4 = (const float4*)w;
  const int HG = 1024;  // histogram grid

  qk_init<<<1, 256, 0, stream>>>(ws);
  qk_hist<0><<<HG, 256, 0, stream>>>(w4, ws);
  qk_select<<<1, 256, 0, stream>>>(ws, 0);
  qk_hist<1><<<HG, 256, 0, stream>>>(w4, ws);
  qk_select<<<1, 256, 0, stream>>>(ws, 1);
  qk_hist<2><<<HG, 256, 0, stream>>>(w4, ws);
  qk_select<<<1, 256, 0, stream>>>(ws, 2);
  qk_hist<3><<<HG, 256, 0, stream>>>(w4, ws);
  qk_select<<<1, 256, 0, stream>>>(ws, 3);
  qk_minabove<<<HG, 256, 0, stream>>>(w4, ws);
  qk_finalize<<<1, 1, 0, stream>>>(ws);

  dim3 grid(OUT_F / TN, BATCH / TM, 1);   // 64 x 4
  gemm_topk<<<grid, 256, 0, stream>>>(x, w, b, ws, out);
}
